// GRU_73134703116294
// MI455X (gfx1250) — matrix operand order repaired and checked by reading, not compiled, not run on hardware
//
#include <hip/hip_runtime.h>
#include <hip/hip_bf16.h>

// GRU forward for MI455X (gfx1250), bf16 WMMA recurrence.
// T=512, B=64, I=1024, H=1024.
// All GEMM operands are pre-packed bf16 (weights once at start; activations
// once per element per step, fused into the epilogues/prologues), so the
// K-loops are pure global_load_b128 + v_wmma_f32_16x16x32_bf16.

#define T_STEPS 512
#define BATCH   64
#define IDIM    1024
#define HDIM    1024
#define GDIM    2048   // 2*H

typedef __attribute__((ext_vector_type(16))) __bf16 v16bf;
typedef __attribute__((ext_vector_type(8)))  float  v8f;
typedef unsigned int   u32;
typedef unsigned short u16;

union Frag {
    u32   u[8];
    uint4 q[2];
    v16bf v;
};

__device__ __forceinline__ u32 pack2_bf16(float lo, float hi) {
    u32 a = __float_as_uint(lo);
    u32 b = __float_as_uint(hi);
    return (a >> 16) | (b & 0xFFFF0000u);
}
__device__ __forceinline__ u16 f32_to_bf16u(float x) {
    return (u16)(__float_as_uint(x) >> 16);
}
__device__ __forceinline__ float sigmoidf_dev(float x) {
    return 1.0f / (1.0f + __expf(-x));
}

// A fragment (16x32 bf16, MxK) from packed-bf16 row (stride in dwords given by
// caller via `row` pointer). Layout matches B: lanes 0-15 hold K=0..15,
// lanes 16-31 hold K=16..31  ->  two contiguous b128 chunks per lane.
__device__ __forceinline__ v16bf load_a_bf16(const u32* __restrict__ row,
                                             int k0dw, int lane) {
    int c0 = k0dw + ((lane & 16) ? 8 : 0);   // +16 bf16 = +8 dwords
    Frag r;
    r.q[0] = *(const uint4*)(row + c0);      // K = c0*2 .. +7
    r.q[1] = *(const uint4*)(row + c0 + 4);  // K = c0*2+8 .. +15
    return r.v;
}

// B fragment (32x16 bf16, KxN): B(k,n) = W[n][k], W row-major (N,K) packed
// bf16. Lanes 0-15: K=0..15 contiguous; lanes 16-31: K=16..31.
__device__ __forceinline__ v16bf load_b_bf16(const u32* __restrict__ row,
                                             int k0dw, int lane) {
    int c0 = k0dw + ((lane & 16) ? 8 : 0);
    Frag r;
    r.q[0] = *(const uint4*)(row + c0);
    r.q[1] = *(const uint4*)(row + c0 + 4);
    return r.v;
}

// -------- generic f32 -> packed bf16 pairs --------
__global__ void gru_f32_to_bf16(const float* __restrict__ src,
                                u32* __restrict__ dst, int n_pairs) {
    int i = blockIdx.x * blockDim.x + threadIdx.x;
    if (i < n_pairs) dst[i] = pack2_bf16(src[2 * i], src[2 * i + 1]);
}

// -------- per-step kernel 1: rz gates --------
// rz = sigmoid(x_t·Wxrz^T + h·Whrz^T + b_rz)
// r-half waves store (r ⊙ h) as bf16 for the next GEMM; z-half waves store z.
// 4 M-tiles x 128 N-tiles = 512 waves; 128 blocks of 4 waves.
__global__ __launch_bounds__(128) void gru_rz_kernel(
    const u32*   __restrict__ xbf,     // (B, I) bf16 packed
    const u32*   __restrict__ hbf,     // (B, H) bf16 packed
    const float* __restrict__ hf,      // (B, H) f32
    const u32*   __restrict__ WxrzBf,  // (2H, I) bf16 packed
    const u32*   __restrict__ WhrzBf,  // (2H, H) bf16 packed
    const float* __restrict__ b_rz,    // (2H)
    u16*         __restrict__ rhbf,    // (B, H) bf16: r*h
    float*       __restrict__ zbuf)    // (B, H) f32: z
{
    const int lane = threadIdx.x & 31;
    const int wave = blockIdx.x * 4 + (threadIdx.x >> 5);
    const int tile_m = wave & 3;       // 64 rows   -> 4 tiles
    const int tile_n = wave >> 2;      // 2048 cols -> 128 tiles
    const int rowA = tile_m * 16 + (lane & 15);
    const int rowB = tile_n * 16 + (lane & 15);

    const u32* xrow = xbf + (size_t)rowA * (IDIM / 2);
    const u32* hrow = hbf + (size_t)rowA * (HDIM / 2);
    const u32* wx   = WxrzBf + (size_t)rowB * (IDIM / 2);
    const u32* wh   = WhrzBf + (size_t)rowB * (HDIM / 2);

    v8f c = {};
    #pragma unroll 4
    for (int k = 0; k < IDIM / 2; k += 16) {   // K步 = 32 elements = 16 dwords
        v16bf a = load_a_bf16(xrow, k, lane);
        v16bf b = load_b_bf16(wx, k, lane);
        c = __builtin_amdgcn_wmma_f32_16x16x32_bf16(false, a, false, b,
                                                    (short)0, c, false, false);
    }
    #pragma unroll 4
    for (int k = 0; k < HDIM / 2; k += 16) {
        v16bf a = load_a_bf16(hrow, k, lane);
        v16bf b = load_b_bf16(wh, k, lane);
        c = __builtin_amdgcn_wmma_f32_16x16x32_bf16(false, a, false, b,
                                                    (short)0, c, false, false);
    }

    const int g = tile_n * 16 + (lane & 15);
    const float bias = b_rz[g];
    const int rbase = tile_m * 16 + ((lane & 16) ? 8 : 0);
    if (tile_n < 64) {                 // r half: g in [0, H)
        #pragma unroll
        for (int i = 0; i < 8; ++i) {
            const int row = rbase + i;
            float r = sigmoidf_dev(c[i] + bias);
            float rh = r * hf[(size_t)row * HDIM + g];
            rhbf[(size_t)row * HDIM + g] = f32_to_bf16u(rh);
        }
    } else {                           // z half: g in [H, 2H)
        const int j = g - HDIM;
        #pragma unroll
        for (int i = 0; i < 8; ++i) {
            const int row = rbase + i;
            zbuf[(size_t)row * HDIM + j] = sigmoidf_dev(c[i] + bias);
        }
    }
}

// -------- per-step kernel 2: candidate + GRU epilogue --------
// gh = tanh(x_t·Wxh^T + (r⊙h)·Whh^T + b_h); h' = (1-z)h + z*gh, masked.
// Also converts x_{t+1} to bf16 for the next step (fused prologue).
// 4 M-tiles x 64 N-tiles = 256 waves; 64 blocks of 4 waves.
__global__ __launch_bounds__(128) void gru_h_kernel(
    const u32*   __restrict__ xbf,      // (B, I) bf16 packed
    const u32*   __restrict__ rhbf,     // (B, H) bf16 packed (r*h)
    const float* __restrict__ hprev,    // (B, H) f32
    const float* __restrict__ zbuf,     // (B, H) f32
    const u32*   __restrict__ WxhBf,    // (H, I) bf16 packed
    const u32*   __restrict__ WhhBf,    // (H, H) bf16 packed
    const float* __restrict__ b_h,      // (H)
    const float* __restrict__ maskT,    // (B)
    const float* __restrict__ Xnext,    // (B, I) f32 of step t+1, or nullptr
    u32*         __restrict__ xbf_next, // (B, I) bf16 packed for step t+1
    float*       __restrict__ hseq,     // d_out + t*B*H
    float*       __restrict__ hf_next,  // (B, H) f32
    u16*         __restrict__ hbf_next) // (B, H) bf16 packed
{
    // fused: convert next step's x tile (64x1024 f32 -> bf16).
    // 64 blocks * 128 threads = 8192 threads * 4 dwords = 32768 dwords.
    if (Xnext != nullptr) {
        const int tid = blockIdx.x * 128 + threadIdx.x;
        const float4* src = (const float4*)Xnext + (size_t)tid * 2;
        float4 f0 = src[0], f1 = src[1];
        uint4 d;
        d.x = pack2_bf16(f0.x, f0.y);  d.y = pack2_bf16(f0.z, f0.w);
        d.z = pack2_bf16(f1.x, f1.y);  d.w = pack2_bf16(f1.z, f1.w);
        ((uint4*)xbf_next)[tid] = d;
    }

    const int lane = threadIdx.x & 31;
    const int wave = blockIdx.x * 4 + (threadIdx.x >> 5);
    const int tile_m = wave & 3;       // 64 rows   -> 4 tiles
    const int tile_n = wave >> 2;      // 1024 cols -> 64 tiles
    const int rowA = tile_m * 16 + (lane & 15);
    const int rowB = tile_n * 16 + (lane & 15);

    const u32* xrow  = xbf  + (size_t)rowA * (IDIM / 2);
    const u32* rhrow = rhbf + (size_t)rowA * (HDIM / 2);
    const u32* wxh   = WxhBf + (size_t)rowB * (IDIM / 2);
    const u32* whh   = WhhBf + (size_t)rowB * (HDIM / 2);

    v8f c = {};
    #pragma unroll 4
    for (int k = 0; k < IDIM / 2; k += 16) {
        v16bf a = load_a_bf16(xrow, k, lane);
        v16bf b = load_b_bf16(wxh, k, lane);
        c = __builtin_amdgcn_wmma_f32_16x16x32_bf16(false, a, false, b,
                                                    (short)0, c, false, false);
    }
    #pragma unroll 4
    for (int k = 0; k < HDIM / 2; k += 16) {
        v16bf a = load_a_bf16(rhrow, k, lane);
        v16bf b = load_b_bf16(whh, k, lane);
        c = __builtin_amdgcn_wmma_f32_16x16x32_bf16(false, a, false, b,
                                                    (short)0, c, false, false);
    }

    const int j = tile_n * 16 + (lane & 15);
    const float bias = b_h[j];
    const int rbase = tile_m * 16 + ((lane & 16) ? 8 : 0);
    #pragma unroll
    for (int i = 0; i < 8; ++i) {
        const int row = rbase + i;
        float gh = tanhf(c[i] + bias);
        float z  = zbuf[(size_t)row * HDIM + j];
        float ho = hprev[(size_t)row * HDIM + j];
        float hn = (1.0f - z) * ho + z * gh;
        float m  = maskT[row];
        hn = hn * m + ho * (1.0f - m);
        hseq   [(size_t)row * HDIM + j] = hn;
        hf_next[(size_t)row * HDIM + j] = hn;
        hbf_next[(size_t)row * HDIM + j] = f32_to_bf16u(hn);
    }
}

extern "C" void kernel_launch(void* const* d_in, const int* in_sizes, int n_in,
                              void* d_out, int out_size, void* d_ws, size_t ws_size,
                              hipStream_t stream) {
    (void)in_sizes; (void)n_in; (void)out_size; (void)ws_size;

    const float* X      = (const float*)d_in[0];  // (T,B,I)
    const float* hidden = (const float*)d_in[1];  // (B,H)
    const float* mask   = (const float*)d_in[2];  // (T,B)
    const float* W_x_rz = (const float*)d_in[3];  // (2H,I)
    const float* W_h_rz = (const float*)d_in[4];  // (2H,H)
    const float* b_rz   = (const float*)d_in[5];  // (2H)
    const float* W_xh   = (const float*)d_in[6];  // (H,I)
    const float* W_hh   = (const float*)d_in[7];  // (H,H)
    const float* b_h    = (const float*)d_in[8];  // (H)
    float* out = (float*)d_out;                   // hiddens (T,B,H) ++ hidden (B,H)

    // ---- workspace layout (~13.5 MB) ----
    u32* Wxrz_bf = (u32*)d_ws;
    u32* Whrz_bf = Wxrz_bf + (size_t)GDIM * (IDIM / 2);
    u32* Wxh_bf  = Whrz_bf + (size_t)GDIM * (HDIM / 2);
    u32* Whh_bf  = Wxh_bf  + (size_t)HDIM * (IDIM / 2);
    u32* xbf0    = Whh_bf  + (size_t)HDIM * (HDIM / 2);
    u32* xbf1    = xbf0 + (size_t)BATCH * (IDIM / 2);
    u32* hbf0    = xbf1 + (size_t)BATCH * (IDIM / 2);
    u32* hbf1    = hbf0 + (size_t)BATCH * (HDIM / 2);
    u32* rhbf    = hbf1 + (size_t)BATCH * (HDIM / 2);
    float* zbuf  = (float*)(rhbf + (size_t)BATCH * (HDIM / 2));
    float* hf0   = zbuf + (size_t)BATCH * HDIM;
    float* hf1   = hf0 + (size_t)BATCH * HDIM;

    // ---- one-time conversions (weights resident in L2 for the scan) ----
    {
        int n;
        n = GDIM * IDIM / 2;
        gru_f32_to_bf16<<<(n + 255) / 256, 256, 0, stream>>>(W_x_rz, Wxrz_bf, n);
        n = GDIM * HDIM / 2;
        gru_f32_to_bf16<<<(n + 255) / 256, 256, 0, stream>>>(W_h_rz, Whrz_bf, n);
        n = HDIM * IDIM / 2;
        gru_f32_to_bf16<<<(n + 255) / 256, 256, 0, stream>>>(W_xh, Wxh_bf, n);
        n = HDIM * HDIM / 2;
        gru_f32_to_bf16<<<(n + 255) / 256, 256, 0, stream>>>(W_hh, Whh_bf, n);
        n = BATCH * IDIM / 2;   // x_0
        gru_f32_to_bf16<<<(n + 255) / 256, 256, 0, stream>>>(X, xbf0, n);
        n = BATCH * HDIM / 2;   // h_0 bf16
        gru_f32_to_bf16<<<(n + 255) / 256, 256, 0, stream>>>(hidden, hbf0, n);
    }
    hipMemcpyAsync(hf0, hidden, (size_t)BATCH * HDIM * sizeof(float),
                   hipMemcpyDeviceToDevice, stream);

    for (int t = 0; t < T_STEPS; ++t) {
        const int p = t & 1;
        u32*   xbf_in  = p ? xbf1 : xbf0;
        u32*   xbf_out = p ? xbf0 : xbf1;
        u32*   hbf_in  = p ? hbf1 : hbf0;
        u32*   hbf_out = p ? hbf0 : hbf1;
        float* hf_in   = p ? hf1 : hf0;
        float* hf_out  = p ? hf0 : hf1;
        const float* Xnext = (t + 1 < T_STEPS)
                           ? X + (size_t)(t + 1) * BATCH * IDIM : nullptr;

        gru_rz_kernel<<<128, 128, 0, stream>>>(
            xbf_in, hbf_in, hf_in, Wxrz_bf, Whrz_bf, b_rz, (u16*)rhbf, zbuf);
        gru_h_kernel<<<64, 128, 0, stream>>>(
            xbf_in, rhbf, hf_in, zbuf, Wxh_bf, Whh_bf, b_h,
            mask + (size_t)t * BATCH, Xnext, xbf_out,
            out + (size_t)t * BATCH * HDIM, hf_out, (u16*)hbf_out);
    }
    // final hidden: t=511 (p=1) wrote hf0
    hipMemcpyAsync(out + (size_t)T_STEPS * BATCH * HDIM, hf0,
                   (size_t)BATCH * HDIM * sizeof(float),
                   hipMemcpyDeviceToDevice, stream);
}
